// NoisyTopkRouter_58463094833555
// MI455X (gfx1250) — compile-verified, matrix operand order checked by reading, not run on hardware
//
#include <hip/hip_runtime.h>
#include <hip/hip_bf16.h>
#include <stdint.h>

// Problem constants (match reference)
#define RT_T     16384
#define RT_D     4096
#define RT_NEXP  128
#define RT_TOPK  8
#define RT_MBLK  32      // tokens per workgroup

typedef __attribute__((ext_vector_type(16))) __bf16 v16bf;
typedef __attribute__((ext_vector_type(8)))  __bf16 v8bf;
typedef __attribute__((ext_vector_type(4)))  __bf16 v4bf;
typedef __attribute__((ext_vector_type(8)))  float  v8f;

// ---------------------------------------------------------------------------
// JAX threefry2x32-20 with key (0,0), matching jax.random.normal(key(0), ...)
// counter layout: for flat index i in [0, n), n even:
//   i <  n/2 : ctr=(i, i+n/2),  take word0
//   i >= n/2 : ctr=(i-n/2, i),  take word1
// ---------------------------------------------------------------------------
__device__ __forceinline__ uint32_t rotl32(uint32_t v, int n) {
    return (v << n) | (v >> (32 - n));
}
__device__ __forceinline__ void tf_round(uint32_t& x0, uint32_t& x1, int r) {
    x0 += x1; x1 = rotl32(x1, r); x1 ^= x0;
}
__device__ __forceinline__ void threefry2x32_key0(uint32_t c0, uint32_t c1,
                                                  uint32_t& o0, uint32_t& o1) {
    const uint32_t k0 = 0u, k1 = 0u, k2 = 0x1BD11BDAu; // k0^k1^parity
    uint32_t x0 = c0 + k0, x1 = c1 + k1;
    tf_round(x0, x1, 13); tf_round(x0, x1, 15); tf_round(x0, x1, 26); tf_round(x0, x1, 6);
    x0 += k1; x1 += k2 + 1u;
    tf_round(x0, x1, 17); tf_round(x0, x1, 29); tf_round(x0, x1, 16); tf_round(x0, x1, 24);
    x0 += k2; x1 += k0 + 2u;
    tf_round(x0, x1, 13); tf_round(x0, x1, 15); tf_round(x0, x1, 26); tf_round(x0, x1, 6);
    x0 += k0; x1 += k1 + 3u;
    tf_round(x0, x1, 17); tf_round(x0, x1, 29); tf_round(x0, x1, 16); tf_round(x0, x1, 24);
    x0 += k1; x1 += k2 + 4u;
    tf_round(x0, x1, 13); tf_round(x0, x1, 15); tf_round(x0, x1, 26); tf_round(x0, x1, 6);
    x0 += k2; x1 += k0 + 5u;
    o0 = x0; o1 = x1;
}

// single-precision erfinv (Giles 2010)
__device__ __forceinline__ float erfinv_f(float x) {
    float w = -__logf((1.0f - x) * (1.0f + x));
    float p;
    if (w < 5.0f) {
        w = w - 2.5f;
        p =            2.81022636e-08f;
        p = fmaf(p, w, 3.43273939e-07f);
        p = fmaf(p, w, -3.5233877e-06f);
        p = fmaf(p, w, -4.39150654e-06f);
        p = fmaf(p, w, 0.00021858087f);
        p = fmaf(p, w, -0.00125372503f);
        p = fmaf(p, w, -0.00417768164f);
        p = fmaf(p, w, 0.246640727f);
        p = fmaf(p, w, 1.50140941f);
    } else {
        w = sqrtf(w) - 3.0f;
        p =            -0.000200214257f;
        p = fmaf(p, w, 0.000100950558f);
        p = fmaf(p, w, 0.00134934322f);
        p = fmaf(p, w, -0.00367342844f);
        p = fmaf(p, w, 0.00573950773f);
        p = fmaf(p, w, -0.0076224613f);
        p = fmaf(p, w, 0.00943887047f);
        p = fmaf(p, w, 1.00167406f);
        p = fmaf(p, w, 2.83297682f);
    }
    return p * x;
}

__device__ __forceinline__ float jax_normal_at(uint32_t i) {
    const uint32_t half = (uint32_t)RT_T * (uint32_t)RT_NEXP / 2u;
    uint32_t c0, c1; bool second;
    if (i < half) { c0 = i;        c1 = i + half; second = false; }
    else          { c0 = i - half; c1 = i;        second = true;  }
    uint32_t o0, o1;
    threefry2x32_key0(c0, c1, o0, o1);
    uint32_t bits = second ? o1 : o0;
    float f01 = __uint_as_float((bits >> 9) | 0x3f800000u) - 1.0f; // [0,1)
    float u = fmaf(f01, 2.0f, -1.0f);                              // [-1,1)
    u = fminf(fmaxf(u, -0.999999940f), 0.999999940f);
    return 1.41421356237f * erfinv_f(u);
}

__device__ __forceinline__ float softplus_f(float x) {
    return fmaxf(x, 0.0f) + log1pf(__expf(-fabsf(x)));
}

// Load a 16-lane-row bf16 WMMA fragment from a row-major f32 row pointer
// (on-the-fly conversion; used for the activation matrix A only).
__device__ __forceinline__ v16bf load_frag_f32cvt(const float* __restrict__ p,
                                                  int off0, int off1) {
    const float4 a0 = *(const float4*)(p + off0);
    const float4 a1 = *(const float4*)(p + off0 + 4);
    const float4 a2 = *(const float4*)(p + off1);
    const float4 a3 = *(const float4*)(p + off1 + 4);
    v16bf f;
    f[0]  = (__bf16)a0.x; f[1]  = (__bf16)a0.y; f[2]  = (__bf16)a0.z; f[3]  = (__bf16)a0.w;
    f[4]  = (__bf16)a1.x; f[5]  = (__bf16)a1.y; f[6]  = (__bf16)a1.z; f[7]  = (__bf16)a1.w;
    f[8]  = (__bf16)a2.x; f[9]  = (__bf16)a2.y; f[10] = (__bf16)a2.z; f[11] = (__bf16)a2.w;
    f[12] = (__bf16)a3.x; f[13] = (__bf16)a3.y; f[14] = (__bf16)a3.z; f[15] = (__bf16)a3.w;
    return f;
}

// Load a bf16 fragment directly from pre-converted weights (no VALU cvt).
__device__ __forceinline__ v16bf load_frag_bf16(const __bf16* __restrict__ p,
                                                int off0, int off1) {
    const v8bf lo = *(const v8bf*)(p + off0);   // 16B
    const v8bf hi = *(const v8bf*)(p + off1);   // 16B
    return __builtin_shufflevector(lo, hi, 0, 1, 2, 3, 4, 5, 6, 7,
                                           8, 9, 10, 11, 12, 13, 14, 15);
}

// ---------------------------------------------------------------------------
// Pre-pass: convert W_gate and W_noise (f32 [128,4096]) to bf16 in d_ws.
// 4 MB read + 2 MB write, once — removes all weight conversions from the
// 512 token-block workgroups of the main kernel.
// ---------------------------------------------------------------------------
__global__ __launch_bounds__(256)
void convert_weights_kernel(const float* __restrict__ Wg,
                            const float* __restrict__ Wn,
                            __bf16* __restrict__ Wgb,
                            __bf16* __restrict__ Wnb) {
    const int i    = blockIdx.x * 256 + threadIdx.x;
    const int base = i * 4;                       // 4 elems per thread per matrix
    const float4 g = *(const float4*)(Wg + base);
    const float4 n = *(const float4*)(Wn + base);
    v4bf gb, nb;
    gb[0] = (__bf16)g.x; gb[1] = (__bf16)g.y; gb[2] = (__bf16)g.z; gb[3] = (__bf16)g.w;
    nb[0] = (__bf16)n.x; nb[1] = (__bf16)n.y; nb[2] = (__bf16)n.z; nb[3] = (__bf16)n.w;
    *(v4bf*)(Wgb + base) = gb;
    *(v4bf*)(Wnb + base) = nb;
}

// ---------------------------------------------------------------------------
// Fused router kernel: per WG -> 32 tokens x all 128 experts.
// 8 waves; wave w: experts [16w, 16w+16).  Per K-step (32): 2 A fragments
// (f32 + cvt) + 1 gate-B + 1 noise-B (bf16 direct) -> 4 WMMAs.
// K-loop unrolled 2x so the scheduler can fill the WMMA->VALU WAR hazard
// slots (4 co-exec) with the other iteration's independent cvt/load work.
// Then: noisy logits -> LDS, barrier, softmax + top-8 per token.
// ---------------------------------------------------------------------------
__global__ __launch_bounds__(256)
void noisy_topk_router_kernel(const float* __restrict__ H,
                              const __bf16* __restrict__ Wgb,
                              const __bf16* __restrict__ Wnb,
                              float* __restrict__ vals_out,
                              int*   __restrict__ inds_out,
                              float* __restrict__ gates_out) {
    __shared__ float sm_logits[RT_MBLK][RT_NEXP]; // 16 KB noisy logits tile

    const int wave = threadIdx.x >> 5;   // 0..7
    const int lane = threadIdx.x & 31;   // wave32
    const int r    = lane & 15;
    const int h    = lane >> 4;
    const int m0   = blockIdx.x * RT_MBLK;  // token block base
    const int n0   = wave * 16;             // expert base for this wave

    const float*  __restrict__ arow0 = H   + (size_t)(m0 + r)      * RT_D;
    const float*  __restrict__ arow1 = H   + (size_t)(m0 + 16 + r) * RT_D;
    const __bf16* __restrict__ grow  = Wgb + (size_t)(n0 + r)      * RT_D;
    const __bf16* __restrict__ nrow  = Wnb + (size_t)(n0 + r)      * RT_D;
    const int off0 = h * 8;
    const int off1 = 16 + h * 8;

    v8f cg0 = {0.f, 0.f, 0.f, 0.f, 0.f, 0.f, 0.f, 0.f};
    v8f cg1 = {0.f, 0.f, 0.f, 0.f, 0.f, 0.f, 0.f, 0.f};
    v8f cn0 = {0.f, 0.f, 0.f, 0.f, 0.f, 0.f, 0.f, 0.f};
    v8f cn1 = {0.f, 0.f, 0.f, 0.f, 0.f, 0.f, 0.f, 0.f};

    #pragma unroll 2
    for (int k0 = 0; k0 < RT_D; k0 += 32) {
        v16bf bg = load_frag_bf16(grow + k0, off0, off1);
        v16bf bn = load_frag_bf16(nrow + k0, off0, off1);
        v16bf a0 = load_frag_f32cvt(arow0 + k0, off0, off1);
        v16bf a1 = load_frag_f32cvt(arow1 + k0, off0, off1);
        cg0 = __builtin_amdgcn_wmma_f32_16x16x32_bf16(
            false, a0, false, bg, (short)0, cg0, false, false);
        cn0 = __builtin_amdgcn_wmma_f32_16x16x32_bf16(
            false, a0, false, bn, (short)0, cn0, false, false);
        cg1 = __builtin_amdgcn_wmma_f32_16x16x32_bf16(
            false, a1, false, bg, (short)0, cg1, false, false);
        cn1 = __builtin_amdgcn_wmma_f32_16x16x32_bf16(
            false, a1, false, bn, (short)0, cn1, false, false);
    }

    // C/D layout: VGPR e: lanes 0-15 -> M=e, N=lane; lanes 16-31 -> M=e+8, N=lane-16
    #pragma unroll
    for (int e = 0; e < 8; ++e) {
        const int m = e + h * 8;
        const int n = n0 + r;
        {   // token tile 0: rows m0 .. m0+15
            const int t = m0 + m;
            const float ns = softplus_f(cn0[e]);
            const float z  = jax_normal_at((uint32_t)(t * RT_NEXP + n));
            sm_logits[m][n] = cg0[e] + z * ns;
        }
        {   // token tile 1: rows m0+16 .. m0+31
            const int t = m0 + 16 + m;
            const float ns = softplus_f(cn1[e]);
            const float z  = jax_normal_at((uint32_t)(t * RT_NEXP + n));
            sm_logits[16 + m][n] = cg1[e] + z * ns;
        }
    }
    __syncthreads();

    // ---- softmax + top-8: wave w handles tokens m = w, w+8, w+16, w+24 ----
    #pragma unroll
    for (int rep = 0; rep < 4; ++rep) {
        const int m = wave + rep * 8;
        const int t = m0 + m;

        float v[4];
        #pragma unroll
        for (int j = 0; j < 4; ++j) v[j] = sm_logits[m][lane + 32 * j];

        float mx = fmaxf(fmaxf(v[0], v[1]), fmaxf(v[2], v[3]));
        #pragma unroll
        for (int off = 16; off > 0; off >>= 1)
            mx = fmaxf(mx, __shfl_xor(mx, off, 32));

        float e4[4], s = 0.0f;
        #pragma unroll
        for (int j = 0; j < 4; ++j) { e4[j] = __expf(v[j] - mx); s += e4[j]; }
        #pragma unroll
        for (int off = 16; off > 0; off >>= 1)
            s += __shfl_xor(s, off, 32);
        const float inv = 1.0f / s;

        float g[4];
        #pragma unroll
        for (int j = 0; j < 4; ++j) {
            g[j] = e4[j] * inv;
            // write-once output, keep out of L2 (weights live there)
            __builtin_nontemporal_store(g[j],
                &gates_out[(size_t)t * RT_NEXP + lane + 32 * j]);
        }

        // iterative argmax: descending values, first-occurrence tie-break
        #pragma unroll
        for (int it = 0; it < RT_TOPK; ++it) {
            float bv = g[0]; int bn2 = lane;
            #pragma unroll
            for (int j = 1; j < 4; ++j) {
                const int n = lane + 32 * j;
                if (g[j] > bv || (g[j] == bv && n < bn2)) { bv = g[j]; bn2 = n; }
            }
            #pragma unroll
            for (int off = 16; off > 0; off >>= 1) {
                const float ov = __shfl_xor(bv, off, 32);
                const int   on = __shfl_xor(bn2, off, 32);
                if (ov > bv || (ov == bv && on < bn2)) { bv = ov; bn2 = on; }
            }
            if (lane == 0) {
                __builtin_nontemporal_store(bv,  &vals_out[(size_t)t * RT_TOPK + it]);
                __builtin_nontemporal_store(bn2, &inds_out[(size_t)t * RT_TOPK + it]);
            }
            #pragma unroll
            for (int j = 0; j < 4; ++j)
                if (bn2 == lane + 32 * j) g[j] = -INFINITY;
        }
    }
}

extern "C" void kernel_launch(void* const* d_in, const int* in_sizes, int n_in,
                              void* d_out, int out_size, void* d_ws, size_t ws_size,
                              hipStream_t stream) {
    (void)in_sizes; (void)n_in; (void)out_size; (void)ws_size;
    const float* H  = (const float*)d_in[0];   // hidden_states [T, D] f32
    const float* Wg = (const float*)d_in[1];   // W_gate [N, D] f32
    const float* Wn = (const float*)d_in[2];   // W_noise [N, D] f32

    float* out   = (float*)d_out;
    float* vals  = out;                                    // [T, 8] f32
    int*   inds  = (int*)(out + (size_t)RT_T * RT_TOPK);   // [T, 8] i32
    float* gates = out + (size_t)2 * RT_T * RT_TOPK;       // [T, 128] f32

    // bf16 weight staging in workspace: 2 x 1 MB
    __bf16* Wgb = (__bf16*)d_ws;
    __bf16* Wnb = Wgb + (size_t)RT_NEXP * RT_D;

    {   // 128*4096 elems / (256 threads * 4 elems) = 512 blocks
        dim3 grid((RT_NEXP * RT_D) / (256 * 4));
        dim3 block(256);
        hipLaunchKernelGGL(convert_weights_kernel, grid, block, 0, stream,
                           Wg, Wn, Wgb, Wnb);
    }
    {
        dim3 grid(RT_T / RT_MBLK);
        dim3 block(256);
        hipLaunchKernelGGL(noisy_topk_router_kernel, grid, block, 0, stream,
                           H, Wgb, Wnb, vals, inds, gates);
    }
}
